// MultiHeadAttention_70549132804907
// MI455X (gfx1250) — compile-verified, matrix operand order checked by reading, not run on hardware
//
#include <hip/hip_runtime.h>
#include <stdint.h>

// ---------------------------------------------------------------------------
// MI455X (gfx1250) multi-head attention: bf16 WMMA pipeline, flash softmax,
// TDM (tensor_load_to_lds) staging of shared GEMM operands, fence-free
// split workgroup barriers.
// ---------------------------------------------------------------------------

typedef __bf16 bf16;
typedef __attribute__((ext_vector_type(16))) __bf16 v16bf;
typedef __attribute__((ext_vector_type(8)))  __bf16 bf16x8;
typedef __attribute__((ext_vector_type(8)))  float  v8f;
typedef __attribute__((ext_vector_type(4)))  unsigned int u32x4;
typedef __attribute__((ext_vector_type(8)))  int i32x8;
typedef __attribute__((ext_vector_type(4)))  int i32x4;

#define N_HEAD 16
#define D_K    64
#define D_MODEL 1024
#define BATCH  2
#define SEQ    2048
#define NEGINF (-1.0e9f)

#ifndef __has_builtin
#define __has_builtin(x) 0
#endif
#if __has_builtin(__builtin_amdgcn_tensor_load_to_lds) && \
    __has_builtin(__builtin_amdgcn_s_wait_tensorcnt)
#define USE_TDM 1
#else
#define USE_TDM 0
#endif

// Fence-free workgroup split barrier: with TDM staging, producer/consumer
// ordering is enforced by s_wait_tensorcnt + the barrier itself, so we skip
// __syncthreads()'s acq_rel fence (which drains LOADcnt and stalls the
// software pipeline). Fallback path keeps the fenced barrier.
#if USE_TDM && __has_builtin(__builtin_amdgcn_s_barrier_signal) && \
    __has_builtin(__builtin_amdgcn_s_barrier_wait)
__device__ inline void wg_barrier() {
  __builtin_amdgcn_s_barrier_signal(-1);
  __builtin_amdgcn_s_barrier_wait(-1);
}
#else
__device__ inline void wg_barrier() { __syncthreads(); }
#endif

// ---------------------------------------------------------------------------
// Fragment loader for 16-bit A/B operands of V_WMMA_F32_16X16X32_BF16.
// lane L holds matrix row (L&15); two contiguous 8-elem (16B) K chunks:
//   halves 0..7  -> K = k0 + 8*(L>=16) + h
//   halves 8..15 -> K = k0 + 16 + 8*(L>=16) + (h-8)
// ---------------------------------------------------------------------------
__device__ inline v16bf load_frag16(const bf16* __restrict__ p, int row0, int ld,
                                    int k0, int lane) {
  const bf16* base = p + (size_t)(row0 + (lane & 15)) * ld + k0 + 8 * (lane >> 4);
  bf16x8 lo = *(const bf16x8*)(base);
  bf16x8 hi = *(const bf16x8*)(base + 16);
  union { v16bf v; bf16x8 h[2]; } u;
  u.h[0] = lo; u.h[1] = hi;
  return u.v;
}

__device__ inline v8f wmma_bf16(v16bf a, v16bf b, v8f c) {
  return __builtin_amdgcn_wmma_f32_16x16x32_bf16(false, a, false, b, (short)0, c,
                                                 false, false);
}

// ---------------------------------------------------------------------------
// TDM 2-D tile load: global (tile_y rows x tile_x contiguous elems, row
// stride src_ld elems, 2-byte data) -> LDS row-major [tile_y][tile_x].
// D# per CDNA5 ISA 8.3/8.4. 6-arg builtin on this toolchain.
// ---------------------------------------------------------------------------
#if USE_TDM
__device__ inline void tdm_load_tile_2d(bf16* dst, const bf16* src,
                                        int tile_x, int tile_y, int src_ld) {
  unsigned lds_off = (unsigned)(unsigned long long)(uintptr_t)(void*)dst;
  unsigned long long ga = (unsigned long long)(uintptr_t)(const void*)src;
  u32x4 g0;
  g0[0] = 1u;                                             // count=1
  g0[1] = lds_off;                                        // lds_addr
  g0[2] = (unsigned)ga;                                   // global_addr lo
  g0[3] = (unsigned)((ga >> 32) & 0x01FFFFFFu) | (2u << 30);  // addr hi | type=2
  i32x8 g1;
  g1[0] = (int)(1u << 16);                                // data_size=2B
  g1[1] = (int)((unsigned)tile_x << 16);                  // tensor_dim0 lo16
  g1[2] = (int)(((unsigned)tile_x >> 16) & 0xFFFFu) |
          (int)(((unsigned)tile_y & 0xFFFFu) << 16);      // dim0 hi | dim1 lo
  g1[3] = (int)(((unsigned)tile_y >> 16) & 0xFFFFu) |
          (int)((unsigned)tile_x << 16);                  // dim1 hi | tile_dim0
  g1[4] = (int)((unsigned)tile_y & 0xFFFFu);              // tile_dim1 (tile_dim2=0)
  g1[5] = (int)(unsigned)src_ld;                          // tensor_dim0_stride lo32
  g1[6] = 0;
  g1[7] = 0;
  i32x4 z4 = {0, 0, 0, 0};
  i32x8 z8 = {0, 0, 0, 0, 0, 0, 0, 0};
  __builtin_amdgcn_tensor_load_to_lds(g0, g1, z4, z4, z8, 0);
}
#endif

// Stage a tile into LDS. TDM path: wave 0 issues one async DMA (TENSORcnt).
// Fallback: all 256 threads cooperative b128 copy.
__device__ inline void stage_tile(bf16* dst, const bf16* src, int tile_x,
                                  int tile_y, int src_ld, int wv) {
#if USE_TDM
  if (wv == 0) tdm_load_tile_2d(dst, src, tile_x, tile_y, src_ld);
#else
  int chunks_per_row = tile_x >> 3;
  int total = tile_y * chunks_per_row;
  for (int i = threadIdx.x; i < total; i += 256) {
    int r = i / chunks_per_row;
    int c = (i - r * chunks_per_row) << 3;
    *(bf16x8*)(dst + r * tile_x + c) = *(const bf16x8*)(src + (size_t)r * src_ld + c);
  }
#endif
}

template <int N>
__device__ inline void stage_wait(int wv) {
#if USE_TDM
  if (wv == 0) __builtin_amdgcn_s_wait_tensorcnt((short)N);
#endif
  (void)wv;
}

// ---------------------------------------------------------------------------
// Kernel 0: dtype conversion + weight transpose to [N][K] bf16.
// ---------------------------------------------------------------------------
__global__ __launch_bounds__(256) void mha_convert_kernel(
    const float* __restrict__ qkv, const float* __restrict__ wq,
    const float* __restrict__ wk, const float* __restrict__ wv,
    const float* __restrict__ wfc, bf16* __restrict__ Xb,
    bf16* __restrict__ Wtq, bf16* __restrict__ Wtk, bf16* __restrict__ Wtv,
    bf16* __restrict__ Wtfc) {
  int job = blockIdx.y;
  size_t idx = (size_t)blockIdx.x * 256 + threadIdx.x;
  size_t stride = (size_t)gridDim.x * 256;
  if (job == 0) {
    size_t n = (size_t)BATCH * SEQ * D_MODEL;
    for (size_t i = idx; i < n; i += stride) Xb[i] = (bf16)qkv[i];
  } else {
    const float* src = (job == 1) ? wq : (job == 2) ? wk : (job == 3) ? wv : wfc;
    bf16* dst = (job == 1) ? Wtq : (job == 2) ? Wtk : (job == 3) ? Wtv : Wtfc;
    size_t n = (size_t)D_MODEL * D_MODEL;
    for (size_t i = idx; i < n; i += stride) {
      size_t nn = i >> 10, kk = i & 1023;
      dst[i] = (bf16)src[kk * D_MODEL + nn];
    }
  }
}

// ---------------------------------------------------------------------------
// Kernel 1: QKV projections; B-panel TDM-staged in LDS (shared by 8 waves),
// A-fragment register double-buffered. One fence-free barrier per k-step:
//   wait(tensorcnt<=0) -> barrier -> issue next tile -> consume current.
// ---------------------------------------------------------------------------
__global__ __launch_bounds__(256) void mha_qkv_proj_kernel(
    const bf16* __restrict__ Xb, const bf16* __restrict__ Wtq,
    const bf16* __restrict__ Wtk, const bf16* __restrict__ Wtv,
    bf16* __restrict__ Qb, bf16* __restrict__ Kb, bf16* __restrict__ Vt) {
  __shared__ alignas(16) bf16 Wbuf[2][64 * 32];  // [n=64][k=32] per buffer
  int lane = threadIdx.x & 31, wv = threadIdx.x >> 5;
  int hi = lane >> 4, ln = lane & 15;
  int which = blockIdx.z;
  const bf16* Wt = (which == 0) ? Wtq : (which == 1) ? Wtk : Wtv;
  int r0 = blockIdx.y * 128 + wv * 16;
  int c0 = blockIdx.x * 64;
  const bf16* Wp0 = Wt + (size_t)c0 * D_MODEL;

  v8f a0 = {}, a1 = {}, a2 = {}, a3 = {};
  stage_tile(Wbuf[0], Wp0, 32, 64, D_MODEL, wv);
  v16bf af = load_frag16(Xb, r0, D_MODEL, 0, lane);
  int buf = 0;
  for (int k = 0; k < D_MODEL; k += 32) {
    int kn = (k + 32) & (D_MODEL - 1);
    stage_wait<0>(wv);
    wg_barrier();
    stage_tile(Wbuf[buf ^ 1], Wp0 + kn, 32, 64, D_MODEL, wv);
    v16bf afn = load_frag16(Xb, r0, D_MODEL, kn, lane);
    const bf16* Wl = Wbuf[buf];
    v16bf b0 = load_frag16(Wl, 0,  32, 0, lane);
    v16bf b1 = load_frag16(Wl, 16, 32, 0, lane);
    v16bf b2 = load_frag16(Wl, 32, 32, 0, lane);
    v16bf b3 = load_frag16(Wl, 48, 32, 0, lane);
    a0 = wmma_bf16(af, b0, a0);
    a1 = wmma_bf16(af, b1, a1);
    a2 = wmma_bf16(af, b2, a2);
    a3 = wmma_bf16(af, b3, a3);
    af = afn;
    buf ^= 1;
  }
  float scale = (which == 0) ? 0.125f : 1.0f;  // 1/sqrt(64) folded into Q
#pragma unroll
  for (int r = 0; r < 8; r++) {
    int g = r0 + r + 8 * hi;
    int bb = g >> 11, ss = g & 2047;
#pragma unroll
    for (int j = 0; j < 4; j++) {
      float v = ((j == 0) ? a0[r] : (j == 1) ? a1[r] : (j == 2) ? a2[r] : a3[r]) * scale;
      int c = c0 + j * 16 + ln;
      int hh = c >> 6, dd = c & 63;
      size_t head = (size_t)(bb * N_HEAD + hh);
      if (which == 2)
        Vt[(head * D_K + dd) * SEQ + ss] = (bf16)v;
      else if (which == 0)
        Qb[(head * SEQ + ss) * D_K + dd] = (bf16)v;
      else
        Kb[(head * SEQ + ss) * D_K + dd] = (bf16)v;
    }
  }
}

// ---------------------------------------------------------------------------
// Kernel 2: flash attention. K/V tiles TDM-staged in LDS, double-buffered,
// shared by all 8 waves (each wave owns 16 q-rows). One barrier per k-block.
// ---------------------------------------------------------------------------
__global__ __launch_bounds__(256) void mha_flash_kernel(
    const bf16* __restrict__ Qb, const bf16* __restrict__ Kb,
    const bf16* __restrict__ Vt, const int* __restrict__ mask,
    bf16* __restrict__ Ob, float* __restrict__ Mrow, float* __restrict__ Lrow) {
  __shared__ alignas(16) bf16 Kbuf[2][32 * 64];  // [key=32][d=64]
  __shared__ alignas(16) bf16 Vbuf[2][64 * 32];  // [d=64][key=32]
  __shared__ alignas(16) bf16 pstage[8][16 * 32];
  int lane = threadIdx.x & 31, wv = threadIdx.x >> 5;
  int hi = lane >> 4, ln = lane & 15;
  int bh = blockIdx.y;
  int b = bh >> 4;
  int q0 = blockIdx.x * 128 + wv * 16;

  const bf16* Qh = Qb + (size_t)bh * SEQ * D_K;
  const bf16* Kh = Kb + (size_t)bh * SEQ * D_K;
  const bf16* Vh = Vt + (size_t)bh * D_K * SEQ;
  const int* mbase = mask + (size_t)b * SEQ * SEQ;

  v16bf aq0 = load_frag16(Qh, q0, D_K, 0, lane);
  v16bf aq1 = load_frag16(Qh, q0, D_K, 32, lane);

  v8f o0 = {}, o1 = {}, o2 = {}, o3 = {};
  float m_r[8], l_r[8];
#pragma unroll
  for (int r = 0; r < 8; r++) { m_r[r] = -3.0e38f; l_r[r] = 0.0f; }

  bf16* pt = pstage[wv];
  stage_tile(Kbuf[0], Kh, 64, 32, 64, wv);
  stage_tile(Vbuf[0], Vh, 32, 64, SEQ, wv);
  int buf = 0;
  for (int kb = 0; kb < SEQ; kb += 32) {
    int nkb = (kb + 32) & (SEQ - 1);
    stage_wait<0>(wv);
    wg_barrier();
    stage_tile(Kbuf[buf ^ 1], Kh + (size_t)nkb * D_K, 64, 32, 64, wv);
    stage_tile(Vbuf[buf ^ 1], Vh + nkb, 32, 64, SEQ, wv);
    const bf16* Kt = Kbuf[buf];
    const bf16* Vl = Vbuf[buf];

    v8f s0 = {}, s1 = {};
    {
      v16bf bk = load_frag16(Kt, 0, D_K, 0, lane);
      s0 = wmma_bf16(aq0, bk, s0);
      bk = load_frag16(Kt, 0, D_K, 32, lane);
      s0 = wmma_bf16(aq1, bk, s0);
      bk = load_frag16(Kt, 16, D_K, 0, lane);
      s1 = wmma_bf16(aq0, bk, s1);
      bk = load_frag16(Kt, 16, D_K, 32, lane);
      s1 = wmma_bf16(aq1, bk, s1);
    }
    float p0[8], p1[8], alpha[8];
#pragma unroll
    for (int r = 0; r < 8; r++) {
      int q = q0 + r + 8 * hi;
      const int* mp = mbase + (size_t)q * SEQ + kb + ln;
      float v0 = mp[0]  ? s0[r] : NEGINF;
      float v1 = mp[16] ? s1[r] : NEGINF;
      float mx = fmaxf(v0, v1);
      mx = fmaxf(mx, __shfl_xor(mx, 1));
      mx = fmaxf(mx, __shfl_xor(mx, 2));
      mx = fmaxf(mx, __shfl_xor(mx, 4));
      mx = fmaxf(mx, __shfl_xor(mx, 8));
      float mn = fmaxf(m_r[r], mx);
      alpha[r] = __expf(m_r[r] - mn);
      m_r[r] = mn;
      p0[r] = __expf(v0 - mn);
      p1[r] = __expf(v1 - mn);
      float sum = p0[r] + p1[r];
      sum += __shfl_xor(sum, 1);
      sum += __shfl_xor(sum, 2);
      sum += __shfl_xor(sum, 4);
      sum += __shfl_xor(sum, 8);
      l_r[r] = l_r[r] * alpha[r] + sum;
    }
#pragma unroll
    for (int r = 0; r < 8; r++) {
      o0[r] *= alpha[r]; o1[r] *= alpha[r];
      o2[r] *= alpha[r]; o3[r] *= alpha[r];
    }
#pragma unroll
    for (int r = 0; r < 8; r++) {
      pt[(r + 8 * hi) * 32 + ln]      = (bf16)p0[r];
      pt[(r + 8 * hi) * 32 + 16 + ln] = (bf16)p1[r];
    }
    v16bf ap = load_frag16(pt, 0, 32, 0, lane);
    v16bf bv;
    bv = load_frag16(Vl, 0,  32, 0, lane); o0 = wmma_bf16(ap, bv, o0);
    bv = load_frag16(Vl, 16, 32, 0, lane); o1 = wmma_bf16(ap, bv, o1);
    bv = load_frag16(Vl, 32, 32, 0, lane); o2 = wmma_bf16(ap, bv, o2);
    bv = load_frag16(Vl, 48, 32, 0, lane); o3 = wmma_bf16(ap, bv, o3);
    buf ^= 1;
  }

  int hh = bh & 15;
#pragma unroll
  for (int r = 0; r < 8; r++) {
    float inv = 1.0f / l_r[r];
    int q = q0 + r + 8 * hi;
    size_t orow = ((size_t)b * SEQ + q) * D_MODEL + hh * D_K;
    Ob[orow + ln]      = (bf16)(o0[r] * inv);
    Ob[orow + 16 + ln] = (bf16)(o1[r] * inv);
    Ob[orow + 32 + ln] = (bf16)(o2[r] * inv);
    Ob[orow + 48 + ln] = (bf16)(o3[r] * inv);
  }
  if (ln == 0) {
#pragma unroll
    for (int r = 0; r < 8; r++) {
      int q = q0 + r + 8 * hi;
      Mrow[(size_t)bh * SEQ + q] = m_r[r];
      Lrow[(size_t)bh * SEQ + q] = l_r[r];
    }
  }
}

// ---------------------------------------------------------------------------
// Kernel 3: materialize attn = exp(s - m)/l (537 MB streaming write; the
// roofline floor of this op). Prefetch next K tile.
// ---------------------------------------------------------------------------
__global__ __launch_bounds__(256) void mha_attn_write_kernel(
    const bf16* __restrict__ Qb, const bf16* __restrict__ Kb,
    const int* __restrict__ mask, const float* __restrict__ Mrow,
    const float* __restrict__ Lrow, float* __restrict__ attn) {
  int lane = threadIdx.x & 31, wv = threadIdx.x >> 5;
  int hi = lane >> 4, ln = lane & 15;
  int bh = blockIdx.y;
  int b = bh >> 4;
  int q0 = blockIdx.x * 16;

  const bf16* Qh = Qb + (size_t)bh * SEQ * D_K;
  const bf16* Kh = Kb + (size_t)bh * SEQ * D_K;
  const int* mbase = mask + (size_t)b * SEQ * SEQ;
  float* aout = attn + (size_t)bh * SEQ * SEQ;

  v16bf aq0 = load_frag16(Qh, q0, D_K, 0, lane);
  v16bf aq1 = load_frag16(Qh, q0, D_K, 32, lane);

  float mv[8], il[8];
#pragma unroll
  for (int r = 0; r < 8; r++) {
    int q = q0 + r + 8 * hi;
    mv[r] = Mrow[(size_t)bh * SEQ + q];
    il[r] = 1.0f / Lrow[(size_t)bh * SEQ + q];
  }

  int kend = wv * 256 + 256;
  for (int kb = wv * 256; kb < kend; kb += 32) {
    __builtin_prefetch(Kh + (size_t)((kb + 32) & (SEQ - 1)) * D_K + lane * 64, 0, 0);
    v8f s0 = {}, s1 = {};
    v16bf bk = load_frag16(Kh, kb, D_K, 0, lane);
    s0 = wmma_bf16(aq0, bk, s0);
    bk = load_frag16(Kh, kb, D_K, 32, lane);
    s0 = wmma_bf16(aq1, bk, s0);
    bk = load_frag16(Kh, kb + 16, D_K, 0, lane);
    s1 = wmma_bf16(aq0, bk, s1);
    bk = load_frag16(Kh, kb + 16, D_K, 32, lane);
    s1 = wmma_bf16(aq1, bk, s1);
#pragma unroll
    for (int r = 0; r < 8; r++) {
      int q = q0 + r + 8 * hi;
      const int* mp = mbase + (size_t)q * SEQ + kb + ln;
      float v0 = mp[0]  ? s0[r] : NEGINF;
      float v1 = mp[16] ? s1[r] : NEGINF;
      size_t row = (size_t)q * SEQ + kb + ln;
      aout[row]      = __expf(v0 - mv[r]) * il[r];
      aout[row + 16] = __expf(v1 - mv[r]) * il[r];
    }
  }
}

// ---------------------------------------------------------------------------
// Kernel 4: output projection + residual (TDM-staged W panel).
// ---------------------------------------------------------------------------
__global__ __launch_bounds__(256) void mha_outproj_kernel(
    const bf16* __restrict__ Ob, const bf16* __restrict__ Wtfc,
    const float* __restrict__ resid, float* __restrict__ out) {
  __shared__ alignas(16) bf16 Wbuf[2][64 * 32];
  int lane = threadIdx.x & 31, wv = threadIdx.x >> 5;
  int hi = lane >> 4, ln = lane & 15;
  int r0 = blockIdx.y * 128 + wv * 16;
  int c0 = blockIdx.x * 64;
  const bf16* Wp0 = Wtfc + (size_t)c0 * D_MODEL;

  v8f a0 = {}, a1 = {}, a2 = {}, a3 = {};
  stage_tile(Wbuf[0], Wp0, 32, 64, D_MODEL, wv);
  v16bf af = load_frag16(Ob, r0, D_MODEL, 0, lane);
  int buf = 0;
  for (int k = 0; k < D_MODEL; k += 32) {
    int kn = (k + 32) & (D_MODEL - 1);
    stage_wait<0>(wv);
    wg_barrier();
    stage_tile(Wbuf[buf ^ 1], Wp0 + kn, 32, 64, D_MODEL, wv);
    v16bf afn = load_frag16(Ob, r0, D_MODEL, kn, lane);
    const bf16* Wl = Wbuf[buf];
    v16bf b0 = load_frag16(Wl, 0,  32, 0, lane);
    v16bf b1 = load_frag16(Wl, 16, 32, 0, lane);
    v16bf b2 = load_frag16(Wl, 32, 32, 0, lane);
    v16bf b3 = load_frag16(Wl, 48, 32, 0, lane);
    a0 = wmma_bf16(af, b0, a0);
    a1 = wmma_bf16(af, b1, a1);
    a2 = wmma_bf16(af, b2, a2);
    a3 = wmma_bf16(af, b3, a3);
    af = afn;
    buf ^= 1;
  }
#pragma unroll
  for (int r = 0; r < 8; r++) {
    int g = r0 + r + 8 * hi;
    size_t row = (size_t)g * D_MODEL;
#pragma unroll
    for (int j = 0; j < 4; j++) {
      int c = c0 + j * 16 + ln;
      float v = (j == 0) ? a0[r] : (j == 1) ? a1[r] : (j == 2) ? a2[r] : a3[r];
      out[row + c] = v + resid[row + c];
    }
  }
}

// ---------------------------------------------------------------------------
// Kernel 5: row-wise LayerNorm in-place on d_out.
// ---------------------------------------------------------------------------
__global__ __launch_bounds__(256) void mha_layernorm_kernel(
    float* __restrict__ out, const float* __restrict__ gamma,
    const float* __restrict__ beta) {
  __shared__ float ssum[8], ssum2[8];
  int lane = threadIdx.x & 31, wv = threadIdx.x >> 5;
  float* p = out + (size_t)blockIdx.x * D_MODEL;
  float x[4], s = 0.0f, s2 = 0.0f;
#pragma unroll
  for (int i = 0; i < 4; i++) {
    x[i] = p[threadIdx.x + i * 256];
    s += x[i];
    s2 += x[i] * x[i];
  }
#pragma unroll
  for (int m = 1; m < 32; m <<= 1) {
    s += __shfl_xor(s, m);
    s2 += __shfl_xor(s2, m);
  }
  if (lane == 0) { ssum[wv] = s; ssum2[wv] = s2; }
  __syncthreads();
  float ts = 0.0f, ts2 = 0.0f;
#pragma unroll
  for (int w = 0; w < 8; w++) { ts += ssum[w]; ts2 += ssum2[w]; }
  float mu = ts * (1.0f / D_MODEL);
  float var = ts2 * (1.0f / D_MODEL) - mu * mu;
  float rs = rsqrtf(var + 1e-6f);
#pragma unroll
  for (int i = 0; i < 4; i++) {
    int c = threadIdx.x + i * 256;
    p[c] = (x[i] - mu) * rs * gamma[c] + beta[c];
  }
}

// ---------------------------------------------------------------------------
// Launch. inputs: [qkv, mask, w_qs, w_ks, w_vs, w_fc, ln_gamma, ln_beta].
// d_out = [out (4096*1024 f32), attn (2*16*2048*2048 f32)].
// ---------------------------------------------------------------------------
extern "C" void kernel_launch(void* const* d_in, const int* in_sizes, int n_in,
                              void* d_out, int out_size, void* d_ws, size_t ws_size,
                              hipStream_t stream) {
  const float* qkv  = (const float*)d_in[0];
  const int*   mask = (const int*)d_in[1];
  const float* wq   = (const float*)d_in[2];
  const float* wk   = (const float*)d_in[3];
  const float* wv   = (const float*)d_in[4];
  const float* wfc  = (const float*)d_in[5];
  const float* gam  = (const float*)d_in[6];
  const float* bet  = (const float*)d_in[7];

  char* ws = (char*)d_ws;
  const size_t MB = 1024ull * 1024ull;
  bf16*  Xb   = (bf16*)(ws);             // 8 MiB  [4096][1024]
  bf16*  Wtq  = (bf16*)(ws + 8 * MB);    // 2 MiB  [N][K]
  bf16*  Wtk  = (bf16*)(ws + 10 * MB);
  bf16*  Wtv  = (bf16*)(ws + 12 * MB);
  bf16*  Wtfc = (bf16*)(ws + 14 * MB);
  bf16*  Qb   = (bf16*)(ws + 16 * MB);   // 8 MiB  [b][h][s][64] (pre-scaled)
  bf16*  Kb   = (bf16*)(ws + 24 * MB);   // 8 MiB  [b][h][s][64]
  bf16*  Vt   = (bf16*)(ws + 32 * MB);   // 8 MiB  [b][h][64][s]
  bf16*  Ob   = (bf16*)(ws + 40 * MB);   // 8 MiB  [b][s][h*64+d]
  float* Mr   = (float*)(ws + 48 * MB);  // 256 KiB
  float* Lr   = (float*)(ws + 48 * MB + 256 * 1024);

  float* out  = (float*)d_out;
  float* attn = out + (size_t)BATCH * SEQ * D_MODEL;

  mha_convert_kernel<<<dim3(2048, 5), 256, 0, stream>>>(
      qkv, wq, wk, wv, wfc, Xb, Wtq, Wtk, Wtv, Wtfc);
  mha_qkv_proj_kernel<<<dim3(16, 32, 3), 256, 0, stream>>>(
      Xb, Wtq, Wtk, Wtv, Qb, Kb, Vt);
  mha_flash_kernel<<<dim3(16, 32), 256, 0, stream>>>(
      Qb, Kb, Vt, mask, Ob, Mr, Lr);
  mha_attn_write_kernel<<<dim3(128, 32), 256, 0, stream>>>(
      Qb, Kb, mask, Mr, Lr, attn);
  mha_outproj_kernel<<<dim3(16, 32), 256, 0, stream>>>(
      Ob, Wtfc, qkv, out);
  mha_layernorm_kernel<<<dim3(4096), 256, 0, stream>>>(out, gam, bet);
}
